// GraphTransformerClassifier_25005299597699
// MI455X (gfx1250) — matrix operand order repaired and checked by reading, not compiled, not run on hardware
//
#include <hip/hip_runtime.h>
#include <hip/hip_bf16.h>
#include <cmath>

// ---------------------------------------------------------------------------
// Types for CDNA5 WMMA (wave32): v_wmma_f32_16x16x32_bf16
// ---------------------------------------------------------------------------
typedef __bf16 bf16;
typedef bf16  v16bf __attribute__((ext_vector_type(16)));
typedef bf16  v8bf  __attribute__((ext_vector_type(8)));
typedef float v8f   __attribute__((ext_vector_type(8)));

constexpr int BATCH  = 2;
constexpr int NPTS   = 2047;
constexpr int DIM    = 512;
constexpr int NH     = 8;
constexpr int FFD    = 2048;
constexpr int NLAYER = 3;
constexpr int NCLS   = 10;
constexpr int PEDIM  = 64;
constexpr int DHD    = DIM / NH;   // 64
constexpr int SEQ    = NPTS + 1;   // 2048
constexpr int MROWS  = BATCH * SEQ; // 4096

// ---------------------------------------------------------------------------
// Fragment helpers (layouts per CDNA5 ISA 7.12.2).
// ---------------------------------------------------------------------------
__device__ inline v16bf cat8(v8bf lo, v8bf hi) {
  return __builtin_shufflevector(lo, hi, 0,1,2,3,4,5,6,7,8,9,10,11,12,13,14,15);
}
__device__ inline v16bf frag_a(const bf16* row, int hf) {
  v8bf lo = *(const v8bf*)(row + hf * 8);
  v8bf hi = *(const v8bf*)(row + 16 + hf * 8);
  return cat8(lo, hi);
}
__device__ inline v16bf frag_b(const bf16* row, int hf) {
  v8bf lo = *(const v8bf*)(row + hf * 8);
  v8bf hi = *(const v8bf*)(row + 16 + hf * 8);
  return cat8(lo, hi);
}
__device__ inline v8f wmma_bf(v16bf a, v16bf b, v8f c) {
  return __builtin_amdgcn_wmma_f32_16x16x32_bf16(false, a, false, b,
                                                 (short)0, c, false, false);
}
__device__ inline float wave_sum32(float v) {
#pragma unroll
  for (int m = 1; m < 32; m <<= 1) v += __shfl_xor(v, m, 32);
  return v;
}

// ---------------------------------------------------------------------------
// CDNA5 async copy: global -> LDS, no VGPR staging, tracked by ASYNCcnt.
// lds_addr: byte offset of destination within the workgroup's LDS
//           (low 32 bits of the flat address of a __shared__ object).
// ---------------------------------------------------------------------------
__device__ inline void async_ld_b128(unsigned lds_addr, const void* gptr) {
  asm volatile("global_load_async_to_lds_b128 %0, %1, off"
               :: "v"(lds_addr), "v"(gptr) : "memory");
}
__device__ inline void wait_async_le3() {
  asm volatile("s_wait_asynccnt 3" ::: "memory");
}
__device__ inline void wait_async_0() {
  asm volatile("s_wait_asynccnt 0" ::: "memory");
}

// ---------------------------------------------------------------------------
// GEMM: C(M,N) = A(M,K)bf16 @ Wt(N,K)bf16^T + bias, with epilogues.
// Block tile 128x64, 8 waves (4x2), each wave 32x32 via 2x2 WMMA tiles, BK=32.
// Tiles are moved with double-buffered GLOBAL_LOAD_ASYNC_TO_LDS_B128:
// issue stage k+1, s_wait_asynccnt 3 (in-order => stage k complete), barrier,
// WMMA on stage k, barrier, swap.
// EPI 0: +bias -> bf16 out | EPI 1: +bias, GELU -> bf16 | EPI 2: +bias+res -> f32
// ---------------------------------------------------------------------------
template<int EPI>
__global__ __launch_bounds__(256) void gemm_bf16_wmma(
    const bf16* __restrict__ A, const bf16* __restrict__ Wt,
    const float* __restrict__ bias, const float* res, void* out,
    int M, int N, int K)
{
  constexpr int STR = 48; // LDS row stride (elements); 96B keeps b128 alignment
  __shared__ __align__(16) bf16 As[2][128 * STR];
  __shared__ __align__(16) bf16 Bs[2][64 * STR];

  const int tid  = threadIdx.x;
  const int lane = tid & 31;
  const int wv   = tid >> 5;
  const int hf   = lane >> 4;
  const int l16  = lane & 15;
  const int mBase = blockIdx.y * 128;
  const int nBase = blockIdx.x * 64;
  const int wm = (wv & 3) * 32;
  const int wn = (wv >> 2) * 32;

  v8f acc[2][2];
#pragma unroll
  for (int i = 0; i < 2; ++i)
#pragma unroll
    for (int j = 0; j < 2; ++j)
#pragma unroll
      for (int e = 0; e < 8; ++e) acc[i][j][e] = 0.f;

  const int ar = tid >> 1, ac = (tid & 1) * 16; // A tile: 128x32, 16 elems/thread
  const int br = tid >> 2, bc = (tid & 3) * 8;  // B tile:  64x32,  8 elems/thread
  const bf16* gA = A  + (size_t)(mBase + ar) * K + ac;
  const bf16* gB = Wt + (size_t)(nBase + br) * K + bc;

  // LDS byte offsets (flat address low 32 bits == LDS offset)
  const unsigned aBuf = (unsigned)(128 * STR * 2);
  const unsigned bBuf = (unsigned)(64 * STR * 2);
  const unsigned aDst = (unsigned)(size_t)(&As[0][0]) + (unsigned)(ar * STR + ac) * 2u;
  const unsigned bDst = (unsigned)(size_t)(&Bs[0][0]) + (unsigned)(br * STR + bc) * 2u;

  // prologue: stage 0 into buffer 0
  async_ld_b128(aDst,      gA);
  async_ld_b128(aDst + 16, gA + 8);
  async_ld_b128(bDst,      gB);

  int cur = 0;
  for (int kb = 0; kb < K; kb += 32) {
    const int nxt = cur ^ 1;
    if (kb + 32 < K) {
      async_ld_b128(aDst + nxt * aBuf,      gA + kb + 32);
      async_ld_b128(aDst + nxt * aBuf + 16, gA + kb + 40);
      async_ld_b128(bDst + nxt * bBuf,      gB + kb + 32);
      wait_async_le3();   // previous batch (3 ops) complete, newest may fly
    } else {
      wait_async_0();
    }
    __syncthreads();

    v16bf af[2], bfr[2];
    af[0]  = frag_a(&As[cur][(wm + l16) * STR], hf);
    af[1]  = frag_a(&As[cur][(wm + 16 + l16) * STR], hf);
    bfr[0] = frag_b(&Bs[cur][(wn + l16) * STR], hf);
    bfr[1] = frag_b(&Bs[cur][(wn + 16 + l16) * STR], hf);
#pragma unroll
    for (int i = 0; i < 2; ++i)
#pragma unroll
      for (int j = 0; j < 2; ++j)
        acc[i][j] = wmma_bf(af[i], bfr[j], acc[i][j]);

    __syncthreads();  // safe to overwrite 'cur' buffer next iteration
    cur = nxt;
  }

#pragma unroll
  for (int i = 0; i < 2; ++i) {
#pragma unroll
    for (int j = 0; j < 2; ++j) {
      const int col = nBase + wn + j * 16 + l16;
      const float bv = bias[col];
#pragma unroll
      for (int r = 0; r < 8; ++r) {
        const int row = mBase + wm + i * 16 + hf * 8 + r;
        const size_t idx = (size_t)row * N + col;
        float v = acc[i][j][r] + bv;
        if constexpr (EPI == 1)
          v = 0.5f * v * (1.f + erff(v * 0.70710678118654752f));
        if constexpr (EPI == 2)
          ((float*)out)[idx] = v + res[idx];
        else
          ((bf16*)out)[idx] = (bf16)v;
      }
    }
  }
}

// ---------------------------------------------------------------------------
// Flash attention. One wave per (b, h, 16-row Q tile). Streams keys in steps
// of 32: 4 WMMAs for S = Q K^T (DH=64), online softmax with width-16 shfl
// reductions, P bounced through LDS (C-layout -> A-layout), 4 WMMAs for P V
// against pre-transposed V (B,H,DH,L). Bias from Kmat with cls row/col = 0.
// ---------------------------------------------------------------------------
__global__ __launch_bounds__(32) void attn_kernel(
    const bf16* __restrict__ Q, const bf16* __restrict__ Kh,
    const bf16* __restrict__ VT, const float* __restrict__ Kmat,
    const float* __restrict__ bias_scale, bf16* __restrict__ O)
{
  __shared__ __align__(16) bf16 pLds[16 * 48];
  const int lane = threadIdx.x & 31;
  const int hf = lane >> 4, l16 = lane & 15;
  const int qBase = blockIdx.x * 16;
  const int h  = blockIdx.y;
  const int bb = blockIdx.z;
  const float bs = bias_scale[h];

  const bf16* qrow = Q + ((size_t)(bb * SEQ + qBase + l16) * DIM) + h * DHD;
  const v16bf qa0 = frag_a(qrow, hf);       // dh 0..31
  const v16bf qa1 = frag_a(qrow + 32, hf);  // dh 32..63

  float rowMax[8], rowSum[8];
  v8f oacc[4];
#pragma unroll
  for (int r = 0; r < 8; ++r) { rowMax[r] = -1e30f; rowSum[r] = 0.f; }
#pragma unroll
  for (int j = 0; j < 4; ++j)
#pragma unroll
    for (int e = 0; e < 8; ++e) oacc[j][e] = 0.f;

  for (int kBase = 0; kBase < SEQ; kBase += 32) {
    const bf16* krow0 = Kh + ((size_t)(bb * SEQ + kBase + l16) * DIM) + h * DHD;
    const bf16* krow1 = Kh + ((size_t)(bb * SEQ + kBase + 16 + l16) * DIM) + h * DHD;
    v8f s0, s1;
#pragma unroll
    for (int e = 0; e < 8; ++e) { s0[e] = 0.f; s1[e] = 0.f; }
    s0 = wmma_bf(qa0, frag_b(krow0, hf), s0);
    s0 = wmma_bf(qa1, frag_b(krow0 + 32, hf), s0);
    s1 = wmma_bf(qa0, frag_b(krow1, hf), s1);
    s1 = wmma_bf(qa1, frag_b(krow1 + 32, hf), s1);

    const int key0 = kBase + l16;
    const int key1 = kBase + 16 + l16;
    float p0[8], p1[8];
#pragma unroll
    for (int r = 0; r < 8; ++r) {
      const int qi = qBase + hf * 8 + r;
      const float b0 = (qi == 0 || key0 == 0)
                           ? 0.f : bs * Kmat[(size_t)(qi - 1) * NPTS + (key0 - 1)];
      const float b1 = (qi == 0 || key1 == 0)
                           ? 0.f : bs * Kmat[(size_t)(qi - 1) * NPTS + (key1 - 1)];
      const float v0 = s0[r] * 0.125f + b0;
      const float v1 = s1[r] * 0.125f + b1;
      float m = fmaxf(v0, v1);
      m = fmaxf(m, __shfl_xor(m, 1, 16));
      m = fmaxf(m, __shfl_xor(m, 2, 16));
      m = fmaxf(m, __shfl_xor(m, 4, 16));
      m = fmaxf(m, __shfl_xor(m, 8, 16));
      const float nm = fmaxf(rowMax[r], m);
      const float corr = __expf(rowMax[r] - nm);
      rowMax[r] = nm;
      p0[r] = __expf(v0 - nm);
      p1[r] = __expf(v1 - nm);
      float ls = p0[r] + p1[r];
      ls += __shfl_xor(ls, 1, 16);
      ls += __shfl_xor(ls, 2, 16);
      ls += __shfl_xor(ls, 4, 16);
      ls += __shfl_xor(ls, 8, 16);
      rowSum[r] = rowSum[r] * corr + ls;
#pragma unroll
      for (int j = 0; j < 4; ++j) oacc[j][r] *= corr;
    }
    // C-layout -> A-layout via LDS (wave-private; DS ops in-order per wave)
#pragma unroll
    for (int r = 0; r < 8; ++r) {
      pLds[(hf * 8 + r) * 48 + l16]      = (bf16)p0[r];
      pLds[(hf * 8 + r) * 48 + 16 + l16] = (bf16)p1[r];
    }
    const v16bf pa = frag_a(&pLds[l16 * 48], hf);
#pragma unroll
    for (int j = 0; j < 4; ++j) {
      const bf16* vrow =
          VT + ((size_t)((bb * NH + h) * DHD + j * 16 + l16) * SEQ) + kBase;
      oacc[j] = wmma_bf(pa, frag_b(vrow, hf), oacc[j]);
    }
  }

#pragma unroll
  for (int j = 0; j < 4; ++j)
#pragma unroll
    for (int r = 0; r < 8; ++r) {
      const int qi = qBase + hf * 8 + r;
      O[((size_t)(bb * SEQ + qi)) * DIM + h * DHD + j * 16 + l16] =
          (bf16)(oacc[j][r] / rowSum[r]);
    }
}

// ---------------------------------------------------------------------------
// LayerNorm over D=512, one wave per row. rowStride lets the final LN pick
// only the two cls rows (rows 0 and 2048) into a compact buffer.
// ---------------------------------------------------------------------------
template<bool F32OUT>
__global__ __launch_bounds__(256) void ln_kernel(
    const float* __restrict__ in, const float* __restrict__ gamma,
    const float* __restrict__ beta, void* __restrict__ out,
    int nRows, int rowStride)
{
  const int wid = blockIdx.x * 8 + (threadIdx.x >> 5);
  if (wid >= nRows) return;
  const int lane = threadIdx.x & 31;
  const float* row = in + (size_t)wid * rowStride * DIM;
  float4 v[4];
  float s = 0.f, ss = 0.f;
#pragma unroll
  for (int i = 0; i < 4; ++i) {
    v[i] = *(const float4*)(row + i * 128 + lane * 4);
    s  += v[i].x + v[i].y + v[i].z + v[i].w;
    ss += v[i].x * v[i].x + v[i].y * v[i].y + v[i].z * v[i].z + v[i].w * v[i].w;
  }
  s = wave_sum32(s);
  ss = wave_sum32(ss);
  const float mean = s * (1.f / DIM);
  const float inv = rsqrtf(ss * (1.f / DIM) - mean * mean + 1e-5f);
#pragma unroll
  for (int i = 0; i < 4; ++i) {
    const float xv[4] = { v[i].x, v[i].y, v[i].z, v[i].w };
#pragma unroll
    for (int c = 0; c < 4; ++c) {
      const int d = i * 128 + lane * 4 + c;
      const float y = (xv[c] - mean) * inv * gamma[d] + beta[d];
      if constexpr (F32OUT)
        ((float*)out)[(size_t)wid * DIM + d] = y;
      else
        ((bf16*)out)[(size_t)wid * DIM + d] = (bf16)y;
    }
  }
}

// ---------------------------------------------------------------------------
// Weight convert: W(K,N) f32 -> Wt(N,K) bf16 (transposed for K-contiguous B).
// ---------------------------------------------------------------------------
__global__ __launch_bounds__(256) void wconv_kernel(
    const float* __restrict__ W, bf16* __restrict__ Wt, int K, int N)
{
  const int idx = blockIdx.x * 256 + threadIdx.x;
  if (idx >= N * K) return;
  const int n = idx / K, k = idx - n * K;
  Wt[idx] = (bf16)W[(size_t)k * N + n];
}

// V(B,L,D) bf16 -> VT(B,H,DH,L) bf16
__global__ __launch_bounds__(256) void vtrans_kernel(
    const bf16* __restrict__ V, bf16* __restrict__ VT)
{
  const int idx = blockIdx.x * 256 + threadIdx.x;
  if (idx >= MROWS * DIM) return;
  const int d = idx & (DIM - 1);
  const int l = (idx >> 9) & (SEQ - 1);
  const int b = idx >> 20;
  const int h = d >> 6, dh = d & 63;
  VT[(((size_t)b * NH + h) * DHD + dh) * SEQ + l] = V[idx];
}

// tok[b,l,d] = l==0 ? cls[d] : x[b,l-1]*W_val[d] + b_val[d] + emb[l-1,d]
//                              + (PE @ W_pe)[l-1,d] + b_pe[d]
__global__ __launch_bounds__(256) void embed_kernel(
    const float* __restrict__ x, const float* __restrict__ PE,
    const float* __restrict__ W_val, const float* __restrict__ b_val,
    const float* __restrict__ emb, const float* __restrict__ W_pe,
    const float* __restrict__ b_pe, const float* __restrict__ clst,
    float* __restrict__ tok)
{
  const int idx = blockIdx.x * 256 + threadIdx.x;
  if (idx >= BATCH * SEQ * DIM) return;
  const int d = idx & (DIM - 1);
  const int l = (idx >> 9) & (SEQ - 1);
  const int b = idx >> 20;
  float v;
  if (l == 0) {
    v = clst[d];
  } else {
    const int pp = l - 1;
    float pe = 0.f;
#pragma unroll 8
    for (int i = 0; i < PEDIM; ++i) pe += PE[pp * PEDIM + i] * W_pe[i * DIM + d];
    v = x[b * NPTS + pp] * W_val[d] + b_val[d] + emb[(size_t)pp * DIM + d]
        + pe + b_pe[d];
  }
  tok[idx] = v;
}

// logits = relu(cls @ Wc1 + bc1) @ Wc2 + bc2 ; cls is (2, 512) fp32.
__global__ __launch_bounds__(256) void head_kernel(
    const float* __restrict__ clsF, const float* __restrict__ Wc1,
    const float* __restrict__ bc1, const float* __restrict__ Wc2,
    const float* __restrict__ bc2, float* __restrict__ out)
{
  __shared__ float hid[DIM / 2];
  const int b = blockIdx.x;
  const int t = threadIdx.x;
  const float* c = clsF + b * DIM;
  float a = bc1[t];
  for (int i = 0; i < DIM; ++i) a += c[i] * Wc1[i * (DIM / 2) + t];
  hid[t] = fmaxf(a, 0.f);
  __syncthreads();
  if (t < NCLS) {
    float o = bc2[t];
    for (int i = 0; i < DIM / 2; ++i) o += hid[i] * Wc2[i * NCLS + t];
    out[b * NCLS + t] = o;
  }
}

// ---------------------------------------------------------------------------
extern "C" void kernel_launch(void* const* d_in, const int* in_sizes, int n_in,
                              void* d_out, int out_size, void* d_ws, size_t ws_size,
                              hipStream_t stream)
{
  (void)in_sizes; (void)n_in; (void)out_size; (void)ws_size;
  const float* x     = (const float*)d_in[0];
  const float* Kmat  = (const float*)d_in[1];
  const float* PE    = (const float*)d_in[2];
  const float* W_val = (const float*)d_in[3];
  const float* b_val = (const float*)d_in[4];
  const float* emb   = (const float*)d_in[5];
  const float* W_pe  = (const float*)d_in[6];
  const float* b_pe  = (const float*)d_in[7];
  const float* clst  = (const float*)d_in[8];
  const float* bscal = (const float*)d_in[9];
  const float* Wq = (const float*)d_in[10];
  const float* bq = (const float*)d_in[11];
  const float* Wk = (const float*)d_in[12];
  const float* bk = (const float*)d_in[13];
  const float* Wv = (const float*)d_in[14];
  const float* bv = (const float*)d_in[15];
  const float* Wo = (const float*)d_in[16];
  const float* bo = (const float*)d_in[17];
  const float* W1 = (const float*)d_in[18];
  const float* b1 = (const float*)d_in[19];
  const float* W2 = (const float*)d_in[20];
  const float* b2 = (const float*)d_in[21];
  const float* g1  = (const float*)d_in[22];
  const float* be1 = (const float*)d_in[23];
  const float* g2  = (const float*)d_in[24];
  const float* be2 = (const float*)d_in[25];
  const float* gf  = (const float*)d_in[26];
  const float* bfv = (const float*)d_in[27];
  const float* Wc1 = (const float*)d_in[28];
  const float* bc1 = (const float*)d_in[29];
  const float* Wc2 = (const float*)d_in[30];
  const float* bc2 = (const float*)d_in[31];

  // Workspace partition (~66 MB total)
  char* p = (char*)d_ws;
  auto alloc = [&](size_t bytes) -> void* {
    void* r = (void*)p;
    p += (bytes + 255) & ~(size_t)255;
    return r;
  };
  float* tok  = (float*)alloc((size_t)MROWS * DIM * 4);
  bf16*  hbf  = (bf16*)alloc((size_t)MROWS * DIM * 2);
  bf16*  Qbf  = (bf16*)alloc((size_t)MROWS * DIM * 2);
  bf16*  Kbf  = (bf16*)alloc((size_t)MROWS * DIM * 2);
  bf16*  Vbf  = (bf16*)alloc((size_t)MROWS * DIM * 2);
  bf16*  VTbf = (bf16*)alloc((size_t)MROWS * DIM * 2);
  bf16*  Obf  = (bf16*)alloc((size_t)MROWS * DIM * 2);
  bf16*  Mid  = (bf16*)alloc((size_t)MROWS * FFD * 2);
  float* clsF = (float*)alloc((size_t)BATCH * DIM * 4);
  bf16 *Wqt[NLAYER], *Wkt[NLAYER], *Wvt[NLAYER], *Wot[NLAYER];
  bf16 *W1t[NLAYER], *W2t[NLAYER];
  for (int l = 0; l < NLAYER; ++l) {
    Wqt[l] = (bf16*)alloc((size_t)DIM * DIM * 2);
    Wkt[l] = (bf16*)alloc((size_t)DIM * DIM * 2);
    Wvt[l] = (bf16*)alloc((size_t)DIM * DIM * 2);
    Wot[l] = (bf16*)alloc((size_t)DIM * DIM * 2);
    W1t[l] = (bf16*)alloc((size_t)FFD * DIM * 2);
    W2t[l] = (bf16*)alloc((size_t)DIM * FFD * 2);
  }

  const int gDD = (DIM * DIM + 255) / 256;
  const int gDF = (DIM * FFD + 255) / 256;
  for (int l = 0; l < NLAYER; ++l) {
    wconv_kernel<<<gDD, 256, 0, stream>>>(Wq + (size_t)l * DIM * DIM, Wqt[l], DIM, DIM);
    wconv_kernel<<<gDD, 256, 0, stream>>>(Wk + (size_t)l * DIM * DIM, Wkt[l], DIM, DIM);
    wconv_kernel<<<gDD, 256, 0, stream>>>(Wv + (size_t)l * DIM * DIM, Wvt[l], DIM, DIM);
    wconv_kernel<<<gDD, 256, 0, stream>>>(Wo + (size_t)l * DIM * DIM, Wot[l], DIM, DIM);
    wconv_kernel<<<gDF, 256, 0, stream>>>(W1 + (size_t)l * DIM * FFD, W1t[l], DIM, FFD);
    wconv_kernel<<<gDF, 256, 0, stream>>>(W2 + (size_t)l * FFD * DIM, W2t[l], FFD, DIM);
  }

  embed_kernel<<<(BATCH * SEQ * DIM + 255) / 256, 256, 0, stream>>>(
      x, PE, W_val, b_val, emb, W_pe, b_pe, clst, tok);

  const dim3 gQKV(DIM / 64, MROWS / 128);  // (8, 32)
  const dim3 gFF1(FFD / 64, MROWS / 128);  // (32, 32)
  const int  gLN = MROWS / 8;              // 512 blocks, wave per row
  const dim3 gATT(SEQ / 16, NH, BATCH);    // (128, 8, 2)
  const int  gEW = (MROWS * DIM + 255) / 256;

  for (int l = 0; l < NLAYER; ++l) {
    ln_kernel<false><<<gLN, 256, 0, stream>>>(tok, g1 + l * DIM, be1 + l * DIM,
                                              hbf, MROWS, 1);
    gemm_bf16_wmma<0><<<gQKV, 256, 0, stream>>>(hbf, Wqt[l], bq + l * DIM,
                                                nullptr, Qbf, MROWS, DIM, DIM);
    gemm_bf16_wmma<0><<<gQKV, 256, 0, stream>>>(hbf, Wkt[l], bk + l * DIM,
                                                nullptr, Kbf, MROWS, DIM, DIM);
    gemm_bf16_wmma<0><<<gQKV, 256, 0, stream>>>(hbf, Wvt[l], bv + l * DIM,
                                                nullptr, Vbf, MROWS, DIM, DIM);
    vtrans_kernel<<<gEW, 256, 0, stream>>>(Vbf, VTbf);
    attn_kernel<<<gATT, 32, 0, stream>>>(Qbf, Kbf, VTbf, Kmat, bscal, Obf);
    gemm_bf16_wmma<2><<<gQKV, 256, 0, stream>>>(Obf, Wot[l], bo + l * DIM,
                                                tok, tok, MROWS, DIM, DIM);
    ln_kernel<false><<<gLN, 256, 0, stream>>>(tok, g2 + l * DIM, be2 + l * DIM,
                                              hbf, MROWS, 1);
    gemm_bf16_wmma<1><<<gFF1, 256, 0, stream>>>(hbf, W1t[l], b1 + l * FFD,
                                                nullptr, Mid, MROWS, FFD, DIM);
    gemm_bf16_wmma<2><<<gQKV, 256, 0, stream>>>(Mid, W2t[l], b2 + l * DIM,
                                                tok, tok, MROWS, DIM, FFD);
  }

  // Final LN only needed for the two cls rows (rows 0 and SEQ).
  ln_kernel<true><<<1, 256, 0, stream>>>(tok, gf, bfv, clsF, BATCH, SEQ);
  head_kernel<<<BATCH, 256, 0, stream>>>(clsF, Wc1, bc1, Wc2, bc2, (float*)d_out);
}
